// GCN_5394478924432
// MI455X (gfx1250) — matrix operand order repaired and checked by reading, not compiled, not run on hardware
//
#include <hip/hip_runtime.h>

#define NN 50000
#define NE 1600000
#define DH 128

typedef __attribute__((ext_vector_type(16))) __bf16 v16bf;
typedef __attribute__((ext_vector_type(8)))  float  v8f;

#define BSTRIDE 20   // dwords per output-column row in LDS (16 pairs + 4 pad; 16B-aligned, conflict-free)

// split fp32 pair into packed bf16 hi / bf16 lo dwords
static __device__ __forceinline__ unsigned pack_split(float x0, float x1, unsigned &loOut) {
    __bf16 h0 = (__bf16)x0, h1 = (__bf16)x1;
    __bf16 l0 = (__bf16)(x0 - (float)h0);
    __bf16 l1 = (__bf16)(x1 - (float)h1);
    union { __bf16 h[2]; unsigned u; } a, b;
    a.h[0] = h0; a.h[1] = h1;
    b.h[0] = l0; b.h[1] = l1;
    loOut = b.u;
    return a.u;
}

// One-time: W[K][128] fp32 -> B-fragment layout: plane[n*(K/2) + p] packed bf16 pairs
__global__ void pack_weights(const float* __restrict__ W, int K,
                             unsigned* __restrict__ hiP, unsigned* __restrict__ loP) {
    int idx = blockIdx.x * 256 + threadIdx.x;
    int total = (K >> 1) * 128;
    if (idx >= total) return;
    int n = idx & 127;                 // coalesced over W columns
    int p = idx >> 7;                  // K pair
    float w0 = W[(size_t)(2 * p) * 128 + n];
    float w1 = W[(size_t)(2 * p + 1) * 128 + n];
    unsigned lo, hi = pack_split(w0, w1, lo);
    size_t o = (size_t)n * (K >> 1) + p;
    hiP[o] = hi;
    loP[o] = lo;
}

// C[N,128] = A[N,K] @ W[K,128] + bias, fp32 via bf16-split WMMA (3 wmma per tile)
__global__ __launch_bounds__(256) void gemm_bias_wmma(
    const float* __restrict__ A, int lda, int K,
    const unsigned* __restrict__ hiP, const unsigned* __restrict__ loP,
    const float* __restrict__ bias,
    float* __restrict__ C, int ldc, int nRows)
{
    extern __shared__ unsigned smem_u[];
    unsigned* sHi = smem_u;                  // [128][BSTRIDE] pairs, column-major
    unsigned* sLo = smem_u + 128 * BSTRIDE;

    const int tid  = threadIdx.x;
    const int lane = tid & 31;
    const int wave = tid >> 5;
    const int m    = lane & 15;
    const int half = lane >> 4;
    const int rowBase = blockIdx.x * 128 + wave * 16;
    int row = rowBase + m;
    if (row >= nRows) row = nRows - 1;                // clamp loads, keep EXEC full
    const float* __restrict__ Arow = A + (size_t)row * lda;
    const int KP = K >> 1;                            // pairs per column

    v8f acc[8];
#pragma unroll
    for (int t = 0; t < 8; ++t)
#pragma unroll
        for (int r = 0; r < 8; ++r) acc[t][r] = 0.0f;

    for (int kc = 0; kc < K; kc += 32) {
        const int kp = kc >> 1;
        // stage pre-packed W chunk into LDS: pure b128 copies, no conversion
#pragma unroll
        for (int q4 = tid; q4 < 512; q4 += 256) {
            int n = q4 >> 2, sub = (q4 & 3) << 2;     // 4 uint4 per column
            size_t go = (size_t)n * KP + kp + sub;
            uint4 h = *reinterpret_cast<const uint4*>(hiP + go);
            uint4 l = *reinterpret_cast<const uint4*>(loP + go);
            *reinterpret_cast<uint4*>(sHi + n * BSTRIDE + sub) = h;
            *reinterpret_cast<uint4*>(sLo + n * BSTRIDE + sub) = l;
        }
        __syncthreads();

        // A fragment (16x32 bf16 layout, ISA 7.12.2): two contiguous 8-float K segments
        union { v16bf v; unsigned u[8]; } aHi, aLo;
        {
            const float* ap = Arow + kc + half * 8;
            float4 a0 = *reinterpret_cast<const float4*>(ap);
            float4 a1 = *reinterpret_cast<const float4*>(ap + 4);
            float4 a2 = *reinterpret_cast<const float4*>(ap + 16);
            float4 a3 = *reinterpret_cast<const float4*>(ap + 20);
            unsigned lo;
            aHi.u[0] = pack_split(a0.x, a0.y, lo); aLo.u[0] = lo;
            aHi.u[1] = pack_split(a0.z, a0.w, lo); aLo.u[1] = lo;
            aHi.u[2] = pack_split(a1.x, a1.y, lo); aLo.u[2] = lo;
            aHi.u[3] = pack_split(a1.z, a1.w, lo); aLo.u[3] = lo;
            aHi.u[4] = pack_split(a2.x, a2.y, lo); aLo.u[4] = lo;
            aHi.u[5] = pack_split(a2.z, a2.w, lo); aLo.u[5] = lo;
            aHi.u[6] = pack_split(a3.x, a3.y, lo); aLo.u[6] = lo;
            aHi.u[7] = pack_split(a3.z, a3.w, lo); aLo.u[7] = lo;
        }

#pragma unroll
        for (int nt = 0; nt < 8; ++nt) {
            const int n = nt * 16 + m;
            // B fragment: pairs half*4 .. half*4+3 and 8+half*4 .. 8+half*4+3 -> 2x ds_load_b128 each
            union { v16bf v; uint4 q[2]; } bHi, bLo;
            const unsigned* bh = sHi + n * BSTRIDE + half * 4;
            const unsigned* bl = sLo + n * BSTRIDE + half * 4;
            bHi.q[0] = *reinterpret_cast<const uint4*>(bh);
            bHi.q[1] = *reinterpret_cast<const uint4*>(bh + 8);
            bLo.q[0] = *reinterpret_cast<const uint4*>(bl);
            bLo.q[1] = *reinterpret_cast<const uint4*>(bl + 8);
            acc[nt] = __builtin_amdgcn_wmma_f32_16x16x32_bf16(false, aHi.v, false, bHi.v,
                                                              (short)0, acc[nt], false, false);
            acc[nt] = __builtin_amdgcn_wmma_f32_16x16x32_bf16(false, aHi.v, false, bLo.v,
                                                              (short)0, acc[nt], false, false);
            acc[nt] = __builtin_amdgcn_wmma_f32_16x16x32_bf16(false, aLo.v, false, bHi.v,
                                                              (short)0, acc[nt], false, false);
        }
        __syncthreads();
    }

    // store: VGPR r -> row rowBase + r + 8*half, col = nt*16 + m
#pragma unroll
    for (int nt = 0; nt < 8; ++nt) {
        int n = nt * 16 + m;
        float bv = bias[n];
#pragma unroll
        for (int r = 0; r < 8; ++r) {
            int rr = rowBase + r + half * 8;
            if (rr < nRows) C[(size_t)rr * ldc + n] = acc[nt][r] + bv;
        }
    }
}

__global__ void fill_zero(float* __restrict__ p, int n) {
    int i = blockIdx.x * 256 + threadIdx.x;
    if (i < n) p[i] = 0.0f;
}

__global__ void deg_count(const int* __restrict__ dst, float* __restrict__ deg, int E) {
    int e = blockIdx.x * 256 + threadIdx.x;
    if (e < E) atomicAdd(&deg[dst[e]], 1.0f);
}

__global__ void dhis_kernel(const float* __restrict__ deg, float* __restrict__ dhis, int n) {
    int i = blockIdx.x * 256 + threadIdx.x;
    if (i < n) dhis[i] = rsqrtf(deg[i] + 1.0f);
}

// v1 = h * dhis^2 (self-loop term), v2 = 0; both streams
__global__ void init_views(const float* __restrict__ ha, const float* __restrict__ hb,
                           const float* __restrict__ dhis,
                           float* __restrict__ v1a, float* __restrict__ v2a,
                           float* __restrict__ v1b, float* __restrict__ v2b, int total) {
    int idx = blockIdx.x * 256 + threadIdx.x;
    if (idx >= total) return;
    int i = idx >> 7;
    float dd = dhis[i]; dd *= dd;
    v1a[idx] = ha[idx] * dd;  v2a[idx] = 0.0f;
    v1b[idx] = hb[idx] * dd;  v2b[idx] = 0.0f;
}

// one wave per edge; lane covers 4 of 128 features; both streams per edge read
__global__ __launch_bounds__(256) void scatter_edges(
    const int* __restrict__ src, const int* __restrict__ dst,
    const float* __restrict__ dhis,
    const float* __restrict__ ha, const float* __restrict__ hb,
    float* __restrict__ v1a, float* __restrict__ v2a,
    float* __restrict__ v1b, float* __restrict__ v2b, int E)
{
    int lane  = threadIdx.x & 31;
    int gwave = (blockIdx.x * blockDim.x + threadIdx.x) >> 5;
    int nwave = (gridDim.x * blockDim.x) >> 5;
    for (int e = gwave; e < E; e += nwave) {
        int s = src[e], d = dst[e];
        float w = dhis[s] * dhis[d];
        size_t so  = (size_t)s * DH + lane * 4;
        size_t dof = (size_t)d * DH + lane * 4;
        float4 a = *reinterpret_cast<const float4*>(ha + so);
        float4 b = *reinterpret_cast<const float4*>(hb + so);
        float* p1a = v1a + dof; float* p2a = v2a + dof;
        float* p1b = v1b + dof; float* p2b = v2b + dof;
        atomicAdd(p1a + 0, a.x * w); atomicAdd(p1a + 1, a.y * w);
        atomicAdd(p1a + 2, a.z * w); atomicAdd(p1a + 3, a.w * w);
        atomicAdd(p2a + 0, a.x);     atomicAdd(p2a + 1, a.y);
        atomicAdd(p2a + 2, a.z);     atomicAdd(p2a + 3, a.w);
        atomicAdd(p1b + 0, b.x * w); atomicAdd(p1b + 1, b.y * w);
        atomicAdd(p1b + 2, b.z * w); atomicAdd(p1b + 3, b.w * w);
        atomicAdd(p2b + 0, b.x);     atomicAdd(p2b + 1, b.y);
        atomicAdd(p2b + 2, b.z);     atomicAdd(p2b + 3, b.w);
    }
}

// q_col = relu(v1a + v2b*Dinv); p_col = relu(v1b + v2a*Dinv); write stride-384 slices
__global__ void combine_relu(const float* __restrict__ v1a, const float* __restrict__ v2a,
                             const float* __restrict__ v1b, const float* __restrict__ v2b,
                             const float* __restrict__ Dinv,
                             float* __restrict__ qOut, float* __restrict__ pOut, int total) {
    int idx = blockIdx.x * 256 + threadIdx.x;
    if (idx >= total) return;
    int i = idx >> 7, f = idx & 127;
    float di = Dinv[i];
    size_t o = (size_t)i * 384 + f;
    qOut[o] = fmaxf(v1a[idx] + v2b[idx] * di, 0.0f);
    pOut[o] = fmaxf(v1b[idx] + v2a[idx] * di, 0.0f);
}

extern "C" void kernel_launch(void* const* d_in, const int* in_sizes, int n_in,
                              void* d_out, int out_size, void* d_ws, size_t ws_size,
                              hipStream_t stream) {
    const float* x     = (const float*)d_in[0];
    const float* view2 = (const float*)d_in[1];
    const int*   ei    = (const int*)d_in[2];
    const float* Dinv  = (const float*)d_in[3];
    const float* Wf[6]  = { (const float*)d_in[4],  (const float*)d_in[6],  (const float*)d_in[8],
                            (const float*)d_in[10], (const float*)d_in[12], (const float*)d_in[14] };
    const float* bf[6]  = { (const float*)d_in[5],  (const float*)d_in[7],  (const float*)d_in[9],
                            (const float*)d_in[11], (const float*)d_in[13], (const float*)d_in[15] };
    const int Kw[6] = { 192, 128, 128, 192, 128, 128 };   // W1,W2,W3,W4,W5,W6

    const int N = NN, E = NE;
    const int* src = ei;
    const int* dst = ei + E;

    float* ws = (float*)d_ws;
    size_t off = 0;
    const size_t nf = (size_t)N * DH;
    float* deg  = ws + off; off += N;
    float* dhis = ws + off; off += N;
    float* ha   = ws + off; off += nf;
    float* hb   = ws + off; off += nf;
    float* v1a  = ws + off; off += nf;
    float* v2a  = ws + off; off += nf;
    float* v1b  = ws + off; off += nf;
    float* v2b  = ws + off; off += nf;
    // packed bf16-split weights (hi/lo planes per weight)
    unsigned* wHi[6]; unsigned* wLo[6];
    unsigned* wsu = (unsigned*)(ws + off);
    size_t uoff = 0;
    for (int i = 0; i < 6; ++i) {
        size_t sz = (size_t)(Kw[i] >> 1) * 128;           // multiple of 16 dwords
        wHi[i] = wsu + uoff; uoff += sz;
        wLo[i] = wsu + uoff; uoff += sz;
    }

    float* q = (float*)d_out;                 // [N,384]
    float* p = q + (size_t)N * 384;           // [N,384]

    const int totNF   = (int)nf;
    const int ewGrid  = 2048;                 // 16384 waves grid-striding edges
    const int ewBlks  = (totNF + 255) / 256;
    const int gemmGrid = (N + 127) / 128;
    const size_t smem  = 2 * 128 * BSTRIDE * sizeof(unsigned);   // 20 KB

    // one-time weight packing
    for (int i = 0; i < 6; ++i) {
        int tot = (Kw[i] >> 1) * 128;
        pack_weights<<<(tot + 255) / 256, 256, 0, stream>>>(Wf[i], Kw[i], wHi[i], wLo[i]);
    }

    // degree -> dhis
    fill_zero<<<(N + 255) / 256, 256, 0, stream>>>(deg, N);
    deg_count<<<(E + 255) / 256, 256, 0, stream>>>(dst, deg, E);
    dhis_kernel<<<(N + 255) / 256, 256, 0, stream>>>(deg, dhis, N);

    struct Lyr { const float* Aa; const float* Ab; int lda; int K; int wa; int wb; int col; };
    Lyr layers[3] = {
        { x,       view2,   192, 192, 0, 3, 0   },    // W1 / W4
        { q,       p,       384, 128, 1, 4, 128 },    // W2 / W5
        { q + 128, p + 128, 384, 128, 2, 5, 256 },    // W3 / W6
    };

    for (int L = 0; L < 3; ++L) {
        const Lyr& ly = layers[L];
        gemm_bias_wmma<<<gemmGrid, 256, smem, stream>>>(ly.Aa, ly.lda, ly.K,
                                                        wHi[ly.wa], wLo[ly.wa], bf[ly.wa], ha, DH, N);
        gemm_bias_wmma<<<gemmGrid, 256, smem, stream>>>(ly.Ab, ly.lda, ly.K,
                                                        wHi[ly.wb], wLo[ly.wb], bf[ly.wb], hb, DH, N);
        init_views<<<ewBlks, 256, 0, stream>>>(ha, hb, dhis, v1a, v2a, v1b, v2b, totNF);
        scatter_edges<<<ewGrid, 256, 0, stream>>>(src, dst, dhis, ha, hb, v1a, v2a, v1b, v2b, E);
        combine_relu<<<ewBlks, 256, 0, stream>>>(v1a, v2a, v1b, v2b, Dinv,
                                                 q + ly.col, p + ly.col, totNF);
    }
}
